// C_Attention_15436112461879
// MI455X (gfx1250) — matrix operand order repaired and checked by reading, not compiled, not run on hardware
//
#include <hip/hip_runtime.h>
#include <hip/hip_bf16.h>

#define DIMC   256
#define HEADS  8
#define NTOK   64
#define HD     32
#define NWIN   64
#define QKV3   768

// LDS row strides (bf16 elements), multiples of 8 for 16B-aligned fragment loads
#define XS 264   // x / attention-out buffer stride (256 + 8)
#define QS 40    // q,k per-head stride (32 + 8)
#define VS 72    // v-transposed per-head stride (64 + 8)
#define PS 72    // probs per-head stride (64 + 8)

__constant__ const float kScale = 0.17677669529663688f; // 32^-0.5

typedef __attribute__((ext_vector_type(16))) __bf16 v16bf;
typedef __attribute__((ext_vector_type(8)))  __bf16 v8bf;
typedef __attribute__((ext_vector_type(4)))  __bf16 v4bf;
typedef __attribute__((ext_vector_type(8)))  float  v8f;

// ---- WMMA helpers -------------------------------------------------------
// A-matrix 16x32 bf16: lane L<16 -> row L, K = {0..7, 16..23}; lane L>=16 -> row L-16, K = {8..15, 24..31}
__device__ __forceinline__ v16bf load_a_frag(const __bf16* p0, int stride, int lane) {
  const int hi = (lane >> 4) << 3;               // 0 or 8
  const __bf16* p = p0 + (lane & 15) * stride + hi;
  union { v16bf v; v8bf h[2]; } u;
  u.h[0] = *(const v8bf*)(p);
  u.h[1] = *(const v8bf*)(p + 16);
  return u.v;
}

// B-matrix 32x16 bf16 (KxN): lane L -> column N = L&15, K = {0..7, 16..23} (L<16) or {8..15, 24..31} (L>=16)
__device__ __forceinline__ v16bf load_b_frag(const __bf16* p0, int stride, int lane) {
  const __bf16* p = p0 + (lane & 15) * stride + ((lane >> 4) << 3);
  union { v16bf v; v8bf h[2]; } u;
  u.h[0] = *(const v8bf*)(p);
  u.h[1] = *(const v8bf*)(p + 16);
  return u.v;
}

__device__ __forceinline__ v8f wmma_bf16(v16bf a, v16bf b, v8f c) {
  return __builtin_amdgcn_wmma_f32_16x16x32_bf16(false, a, false, b, (short)0, c, false, false);
}

// ---- Prep: bf16 weight conversion + fused (relative-bias + window-mask) --
__global__ void swin_prep_kernel(const float* __restrict__ qkv_w,
                                 const float* __restrict__ proj_w,
                                 const float* __restrict__ mask,
                                 const float* __restrict__ bias_table,
                                 const int*   __restrict__ rl_ind,
                                 __bf16* __restrict__ qkv_wb,
                                 __bf16* __restrict__ proj_wb,
                                 float*  __restrict__ combo) {
  int i = blockIdx.x * 256 + threadIdx.x;             // grid covers 2,097,152
  if (i < QKV3 * DIMC) qkv_wb[i] = (__bf16)qkv_w[i];
  if (i < DIMC * DIMC) proj_wb[i] = (__bf16)proj_w[i];
  // combo[w][h][n][m] = mask[w][n][m] + bias_table[rl_ind[n][m]][h]
  int m = i & 63;
  int n = (i >> 6) & 63;
  int h = (i >> 12) & 7;
  int w = i >> 15;
  combo[i] = mask[(w * NTOK + n) * NTOK + m] + bias_table[rl_ind[n * NTOK + m] * HEADS + h];
}

// ---- Fused window attention: 1 block = 1 window, 256 threads = 8 waves --
__global__ __launch_bounds__(256, 1)
void swin_attn_kernel(const float* __restrict__ x,
                      const float* __restrict__ qkv_b,
                      const float* __restrict__ proj_b,
                      const __bf16* __restrict__ qkv_wb,
                      const __bf16* __restrict__ proj_wb,
                      const float* __restrict__ combo,
                      float* __restrict__ out) {
  __shared__ __align__(32) __bf16 sX [NTOK * XS];          // x bf16, reused for attn-out
  __shared__ __align__(32) __bf16 sQ [HEADS * NTOK * QS];  // q (pre-scaled)
  __shared__ __align__(32) __bf16 sK [HEADS * NTOK * QS];  // k
  __shared__ __align__(32) __bf16 sVT[HEADS * HD   * VS];  // v transposed [d][m]
  __shared__ __align__(32) __bf16 sP [HEADS * NTOK * PS];  // softmax probs

  const int b     = blockIdx.x;
  const int tid   = threadIdx.x;
  const int lane  = tid & 31;
  const int wave  = tid >> 5;
  const int widx  = b & (NWIN - 1);
  const int lcol  = lane & 15;           // N-coordinate within a tile
  const int rbase = (lane >> 4) << 3;    // M-row base (0 or 8) for C/D fragments

  // --- Stage 1: x (fp32, HBM) -> bf16 LDS -------------------------------
  {
    const float4* xg = (const float4*)(x + (size_t)b * NTOK * DIMC);
    #pragma unroll
    for (int it = 0; it < 16; ++it) {
      int p   = it * 256 + tid;          // 0..4095 float4s
      int row = p >> 6;
      int c4  = p & 63;
      float4 f = xg[p];
      v4bf t = { (__bf16)f.x, (__bf16)f.y, (__bf16)f.z, (__bf16)f.w };
      *(v4bf*)(&sX[row * XS + c4 * 4]) = t;
    }
  }
  __syncthreads();

  // --- Stage 2: qkv = x @ qkv_w^T + b  (64 x 768 x 256) -----------------
  for (int jt = wave * 6; jt < wave * 6 + 6; ++jt) {
    const int j0 = jt * 16;
    v8f acc[4] = {};
    for (int kt = 0; kt < 8; ++kt) {
      const int k0 = kt * 32;
      v16bf bf = load_b_frag(qkv_wb + (size_t)j0 * DIMC + k0, DIMC, lane);
      #pragma unroll
      for (int mt = 0; mt < 4; ++mt) {
        v16bf af = load_a_frag(sX + mt * 16 * XS + k0, XS, lane);
        acc[mt] = wmma_bf16(af, bf, acc[mt]);
      }
    }
    const int jc  = j0 + lcol;           // global qkv column (0..767), sec uniform per tile
    const int sec = jc >> 8;             // 0=q 1=k 2=v
    const int jj  = jc & 255;
    const int h   = jj >> 5;
    const int d   = jj & 31;
    const float bias = qkv_b[jc];
    #pragma unroll
    for (int mt = 0; mt < 4; ++mt) {
      #pragma unroll
      for (int vr = 0; vr < 8; ++vr) {
        const int n = mt * 16 + rbase + vr;
        const float val = acc[mt][vr] + bias;
        if (sec == 0)      sQ[(h * NTOK + n) * QS + d] = (__bf16)(val * kScale);
        else if (sec == 1) sK[(h * NTOK + n) * QS + d] = (__bf16)val;
        else               sVT[(h * HD + d) * VS + n]  = (__bf16)val;   // transpose
      }
    }
  }
  __syncthreads();

  // --- Stage 3: per-head attention (wave w -> head w) -------------------
  {
    const int h = wave;
    const __bf16* qh = sQ  + h * NTOK * QS;
    const __bf16* kh = sK  + h * NTOK * QS;
    const __bf16* vh = sVT + h * HD   * VS;
    __bf16*       ph = sP  + h * NTOK * PS;

    // logits = (q*scale) @ k^T : K=32 is a single WMMA step
    v8f c[4][4];
    #pragma unroll
    for (int mt = 0; mt < 4; ++mt) {
      v16bf af = load_a_frag(qh + mt * 16 * QS, QS, lane);
      #pragma unroll
      for (int nt = 0; nt < 4; ++nt) {
        v16bf bf = load_b_frag(kh + nt * 16 * QS, QS, lane);
        v8f z = {};
        c[mt][nt] = wmma_bf16(af, bf, z);
      }
    }

    // fused bias+mask add, then in-register softmax (rows live in 16-lane halves)
    const float* cb = combo + (size_t)(widx * HEADS + h) * NTOK * NTOK;
    #pragma unroll
    for (int mt = 0; mt < 4; ++mt) {
      #pragma unroll
      for (int vr = 0; vr < 8; ++vr) {
        const int n = mt * 16 + rbase + vr;
        const float* cbn = cb + n * NTOK + lcol;
        float v0 = c[mt][0][vr] + cbn[0];
        float v1 = c[mt][1][vr] + cbn[16];
        float v2 = c[mt][2][vr] + cbn[32];
        float v3 = c[mt][3][vr] + cbn[48];
        float mx = fmaxf(fmaxf(v0, v1), fmaxf(v2, v3));
        #pragma unroll
        for (int s = 1; s < 16; s <<= 1) mx = fmaxf(mx, __shfl_xor(mx, s, 32));
        float e0 = __expf(v0 - mx), e1 = __expf(v1 - mx);
        float e2 = __expf(v2 - mx), e3 = __expf(v3 - mx);
        float sum = e0 + e1 + e2 + e3;
        #pragma unroll
        for (int s = 1; s < 16; s <<= 1) sum += __shfl_xor(sum, s, 32);
        const float inv = 1.0f / sum;
        __bf16* pr = ph + n * PS + lcol;
        pr[0]  = (__bf16)(e0 * inv);
        pr[16] = (__bf16)(e1 * inv);
        pr[32] = (__bf16)(e2 * inv);
        pr[48] = (__bf16)(e3 * inv);
      }
    }

    // out_h = P @ V : M=64, N=32, K=64 (B-frags contiguous thanks to v^T layout)
    v8f o[4][2] = {};
    #pragma unroll
    for (int kt = 0; kt < 2; ++kt) {
      #pragma unroll
      for (int ntd = 0; ntd < 2; ++ntd) {
        v16bf bf = load_b_frag(vh + ntd * 16 * VS + kt * 32, VS, lane);
        #pragma unroll
        for (int mt = 0; mt < 4; ++mt) {
          v16bf af = load_a_frag(ph + mt * 16 * PS + kt * 32, PS, lane);
          o[mt][ntd] = wmma_bf16(af, bf, o[mt][ntd]);
        }
      }
    }

    // write per-head output into sX (x no longer needed) as bf16 [n][h*32+d]
    #pragma unroll
    for (int mt = 0; mt < 4; ++mt)
      #pragma unroll
      for (int ntd = 0; ntd < 2; ++ntd)
        #pragma unroll
        for (int vr = 0; vr < 8; ++vr) {
          const int n   = mt * 16 + rbase + vr;
          const int col = h * HD + ntd * 16 + lcol;
          sX[n * XS + col] = (__bf16)(o[mt][ntd][vr]);
        }
  }
  __syncthreads();

  // --- Stage 4: proj GEMM (64 x 256 x 256) + bias -> HBM fp32 -----------
  for (int jt = wave * 2; jt < wave * 2 + 2; ++jt) {
    const int j0 = jt * 16;
    v8f acc[4] = {};
    for (int kt = 0; kt < 8; ++kt) {
      const int k0 = kt * 32;
      v16bf bf = load_b_frag(proj_wb + (size_t)j0 * DIMC + k0, DIMC, lane);
      #pragma unroll
      for (int mt = 0; mt < 4; ++mt) {
        v16bf af = load_a_frag(sX + mt * 16 * XS + k0, XS, lane);
        acc[mt] = wmma_bf16(af, bf, acc[mt]);
      }
    }
    const int jc = j0 + lcol;
    const float pb = proj_b[jc];
    float* og = out + (size_t)b * NTOK * DIMC;
    #pragma unroll
    for (int mt = 0; mt < 4; ++mt)
      #pragma unroll
      for (int vr = 0; vr < 8; ++vr) {
        const int n = mt * 16 + rbase + vr;
        og[n * DIMC + jc] = acc[mt][vr] + pb;
      }
  }
}

extern "C" void kernel_launch(void* const* d_in, const int* in_sizes, int n_in,
                              void* d_out, int out_size, void* d_ws, size_t ws_size,
                              hipStream_t stream) {
  const float* x          = (const float*)d_in[0];
  const float* mask       = (const float*)d_in[1];
  const float* qkv_w      = (const float*)d_in[2];
  const float* qkv_b      = (const float*)d_in[3];
  const float* proj_w     = (const float*)d_in[4];
  const float* proj_b     = (const float*)d_in[5];
  const float* bias_table = (const float*)d_in[6];
  const int*   rl_ind     = (const int*)d_in[7];
  float* out = (float*)d_out;

  // workspace: bf16 weights + fused bias+mask table (L2-resident, ~8.9 MB)
  char* ws = (char*)d_ws;
  __bf16* qkv_wb  = (__bf16*)ws;                                   // 768*256*2 = 393,216 B
  __bf16* proj_wb = (__bf16*)(ws + 393216);                        // 256*256*2 = 131,072 B
  float*  combo   = (float*)(ws + 393216 + 131072);                // 64*8*64*64*4 = 8 MB

  swin_prep_kernel<<<8192, 256, 0, stream>>>(qkv_w, proj_w, mask, bias_table, rl_ind,
                                             qkv_wb, proj_wb, combo);
  swin_attn_kernel<<<4096, 256, 0, stream>>>(x, qkv_b, proj_b, qkv_wb, proj_wb,
                                             combo, out);
}
